// Quant_dwpw2_34617436405816
// MI455X (gfx1250) — matrix operand order repaired and checked by reading, not compiled, not run on hardware
//
#include <hip/hip_runtime.h>
#include <hip/hip_bf16.h>

#define LEAK 0.1f
#define BN_EPS 1e-5f
#define KCB 512   // all feature codebooks have 512 entries

#define AS1 __attribute__((address_space(1)))
#define AS3 __attribute__((address_space(3)))

#if __has_builtin(__builtin_amdgcn_global_load_async_to_lds_b128)
#define HAVE_ASYNC_LDS 1
#endif

typedef __attribute__((ext_vector_type(16))) __bf16 bf16x16;
typedef __attribute__((ext_vector_type(8)))  float  f32x8;
typedef int v4i __attribute__((vector_size(4 * sizeof(int))));

// f32 -> bf16 round-to-nearest-even, bit level
__device__ __forceinline__ unsigned short f2bf(float f) {
  unsigned int u = __float_as_uint(f);
  u += 0x7FFFu + ((u >> 16) & 1u);
  return (unsigned short)(u >> 16);
}

__device__ __forceinline__ void wait_async0() {
#ifdef HAVE_ASYNC_LDS
#if __has_builtin(__builtin_amdgcn_s_wait_asynccnt)
  __builtin_amdgcn_s_wait_asynccnt(0);
#else
  asm volatile("s_wait_asynccnt 0" ::: "memory");
#endif
#endif
}

union BfFrag { bf16x16 v; unsigned short u[16]; };

// A fragment (16x32 bf16, ISA 7.12.2 "16-bit A-Matrix 16x32"):
// lanes 0-15: m=lane, kb=0 ; lanes 16-31: m=lane-16, kb=8
// vector elem j -> K = kb + j + (j>=8 ? 8 : 0)
__device__ __forceinline__ bf16x16 frag_a(const unsigned short* tile, int lane) {
  const unsigned short* row = tile + (lane & 15) * 32;
  const int kb = (lane < 16) ? 0 : 8;
  BfFrag f;
#pragma unroll
  for (int j = 0; j < 16; ++j) f.u[j] = row[kb + j + ((j >= 8) ? 8 : 0)];
  return f.v;
}

// B fragment (32x16). Tile stored [col][32 k] in LDS (codebook-row chunks).
// lanes 0-15: n=lane, kb=0 ; lanes 16-31: n=lane-16, kb=8; elem j -> K = kb + j + (j>=8 ? 8 : 0)
__device__ __forceinline__ bf16x16 frag_b(const unsigned short* tile, int lane) {
  const unsigned short* col = tile + (lane & 15) * 32;
  const int kb = (lane < 16) ? 0 : 8;
  BfFrag f;
#pragma unroll
  for (int j = 0; j < 16; ++j) f.u[j] = col[kb + j + ((j >= 8) ? 8 : 0)];
  return f.v;
}

// Stage one 512x32 bf16 codebook chunk (64 B per row) into LDS.
// CB is pre-converted, padded bf16 (row stride Dp, multiple of 32): pure copy.
__device__ __forceinline__ void stage_b(unsigned short* lds_b,
                                        const unsigned short* __restrict__ CB,
                                        int Dp, int k0, int tid) {
#ifdef HAVE_ASYNC_LDS
#pragma unroll
  for (int it = 0; it < 8; ++it) {
    const int q = tid + it * 256;          // 0..2047 = 512 rows * 4 segs
    const int row = q >> 2, seg = q & 3;
    const unsigned short* g = CB + (long)row * Dp + k0 + seg * 8;
    unsigned short* l = lds_b + row * 32 + seg * 8;
    __builtin_amdgcn_global_load_async_to_lds_b128(
        (AS1 v4i*)g, (AS3 v4i*)l, 0, 0);
  }
#else
#pragma unroll
  for (int it = 0; it < 8; ++it) {
    const int q = tid + it * 256;
    const int row = q >> 2, seg = q & 3;
    const uint4 v = *(const uint4*)(CB + (long)row * Dp + k0 + seg * 8);
    *(uint4*)(lds_b + row * 32 + seg * 8) = v;
  }
#endif
}

// ---------------------------------------------------------------------------
// Generic VQ scorer: block handles 16 rows of X vs all 512 codebook rows.
// 8 waves * 4 col-tiles = 512 cols. bf16 WMMA, f32 accumulate.
// idx_out != null : per-row argmin of (||cb||^2 - 2<x,cb>)  (== argmin dist)
// s_out   != null : store raw dot products (Gram-matrix mode)
// ---------------------------------------------------------------------------
__global__ __launch_bounds__(256)
void vq_gemm_kernel(const float* __restrict__ X, int ldx,
                    const unsigned short* __restrict__ CB, int Dp,
                    const float* __restrict__ cbnorm,
                    int N, int D,
                    int* __restrict__ idx_out, float* __restrict__ s_out) {
  __shared__ __align__(16) unsigned short lds_a[16 * 32];
  __shared__ __align__(16) unsigned short lds_b[KCB * 32];
  __shared__ float red_v[16][16];
  __shared__ int   red_i[16][16];

  const int tid  = threadIdx.x;
  const int lane = tid & 31;
  const int wave = tid >> 5;               // 0..7
  const int rowbase = blockIdx.x * 16;

  f32x8 acc[4] = {};

  const int nchunks = Dp >> 5;
  for (int kc = 0; kc < nchunks; ++kc) {
    const int k0 = kc << 5;
    // stage A: 16 rows x 32 k (f32 -> bf16)
    for (int i = tid; i < 16 * 32; i += 256) {
      const int r = i >> 5, kk = k0 + (i & 31);
      const int row = rowbase + r;
      float v = (row < N && kk < D) ? X[(long)row * ldx + kk] : 0.0f;
      lds_a[i] = f2bf(v);
    }
    // stage B: 512 codebook rows x 32 k (pure bf16 copy / async DMA)
    stage_b(lds_b, CB, Dp, k0, tid);
    wait_async0();
    __syncthreads();
    const bf16x16 a = frag_a(lds_a, lane);
#pragma unroll
    for (int t = 0; t < 4; ++t) {
      const bf16x16 b = frag_b(lds_b + (wave * 64 + t * 16) * 32, lane);
      acc[t] = __builtin_amdgcn_wmma_f32_16x16x32_bf16(
          false, a, false, b, (short)0, acc[t], false, false);
    }
    __syncthreads();
  }

  if (s_out) {  // Gram mode: raw dot products
#pragma unroll
    for (int t = 0; t < 4; ++t) {
      const int col = wave * 64 + t * 16 + (lane & 15);
#pragma unroll
      for (int r = 0; r < 8; ++r) {
        const int row = rowbase + r + ((lane < 16) ? 0 : 8);
        if (row < N) s_out[(long)row * KCB + col] = acc[t][r];
      }
    }
    return;
  }

  // argmin mode: maximize (dot - 0.5*||cb||^2). Reuse lds_b as [16][512] f32.
  float* score = (float*)lds_b;   // 16*512*4 == sizeof(lds_b)
#pragma unroll
  for (int t = 0; t < 4; ++t) {
    const int col = wave * 64 + t * 16 + (lane & 15);
    const float adj = 0.5f * cbnorm[col];
#pragma unroll
    for (int r = 0; r < 8; ++r) {
      const int row = r + ((lane < 16) ? 0 : 8);
      score[row * KCB + col] = acc[t][r] - adj;
    }
  }
  __syncthreads();
  {
    const int row = tid >> 4, seg = tid & 15;
    float best = -3.4e38f; int bi = 0;
    for (int c0 = seg * 32; c0 < seg * 32 + 32; ++c0) {
      const float v = score[row * KCB + c0];
      if (v > best) { best = v; bi = c0; }
    }
    red_v[row][seg] = best; red_i[row][seg] = bi;
  }
  __syncthreads();
  if (tid < 16) {
    float best = -3.4e38f; int bi = 0;
    for (int s = 0; s < 16; ++s) {
      const float v = red_v[tid][s];
      if (v > best) { best = v; bi = red_i[tid][s]; }
    }
    const int row = rowbase + tid;
    if (row < N) idx_out[row] = bi;
  }
}

// ---------------------------------------------------------------------------
// Fused conv2 + VQ: rows (b,g,c) are depthwise-conv outputs of gathered
// cb_fdw2 codewords; conv is done during A-tile staging (never materialized).
// ---------------------------------------------------------------------------
__global__ __launch_bounds__(256)
void conv_vq_kernel(const float* __restrict__ V,      // cb_fdw2 [512][2116] = 46x46
                    const int* __restrict__ idx3,
                    const float* __restrict__ qw2,    // [32][9]
                    const unsigned short* __restrict__ CB, // bf16 cb_fpw2 [512][1952]
                    const float* __restrict__ cbnorm, // [512]
                    int* __restrict__ idx_out) {
  __shared__ __align__(16) unsigned short lds_a[16 * 32];
  __shared__ __align__(16) unsigned short lds_b[KCB * 32];
  __shared__ float red_v[16][16];
  __shared__ int   red_i[16][16];

  const int tid  = threadIdx.x;
  const int lane = tid & 31;
  const int wave = tid >> 5;
  const int rowbase = blockIdx.x * 16;
  const int D  = 1936;                     // 44*44
  const int Dp = 1952;                     // padded

  f32x8 acc[4] = {};

  const int nchunks = Dp >> 5;             // 61
  for (int kc = 0; kc < nchunks; ++kc) {
    const int k0 = kc << 5;
    // stage A: compute 3x3 valid-conv outputs on the fly
    for (int i = tid; i < 16 * 32; i += 256) {
      const int r = i >> 5;
      const int p = k0 + (i & 31);
      float v = 0.0f;
      if (p < D) {
        const int n = rowbase + r;
        const int j = idx3[n];
        const int g = (n >> 5) & 31;
        const float* vr = V + (long)j * 2116;
        const float* f  = qw2 + g * 9;
        const int y = p / 44, x = p - y * 44;
        const float* s = vr + y * 46 + x;
        v = s[0]*f[0]  + s[1]*f[1]  + s[2]*f[2]
          + s[46]*f[3] + s[47]*f[4] + s[48]*f[5]
          + s[92]*f[6] + s[93]*f[7] + s[94]*f[8];
      }
      lds_a[i] = f2bf(v);
    }
    stage_b(lds_b, CB, Dp, k0, tid);
    wait_async0();
    __syncthreads();
    const bf16x16 a = frag_a(lds_a, lane);
#pragma unroll
    for (int t = 0; t < 4; ++t) {
      const bf16x16 b = frag_b(lds_b + (wave * 64 + t * 16) * 32, lane);
      acc[t] = __builtin_amdgcn_wmma_f32_16x16x32_bf16(
          false, a, false, b, (short)0, acc[t], false, false);
    }
    __syncthreads();
  }

  float* score = (float*)lds_b;
#pragma unroll
  for (int t = 0; t < 4; ++t) {
    const int col = wave * 64 + t * 16 + (lane & 15);
    const float adj = 0.5f * cbnorm[col];
#pragma unroll
    for (int r = 0; r < 8; ++r) {
      const int row = r + ((lane < 16) ? 0 : 8);
      score[row * KCB + col] = acc[t][r] - adj;
    }
  }
  __syncthreads();
  {
    const int row = tid >> 4, seg = tid & 15;
    float best = -3.4e38f; int bi = 0;
    for (int c0 = seg * 32; c0 < seg * 32 + 32; ++c0) {
      const float v = score[row * KCB + c0];
      if (v > best) { best = v; bi = c0; }
    }
    red_v[row][seg] = best; red_i[row][seg] = bi;
  }
  __syncthreads();
  if (tid < 16) {
    float best = -3.4e38f; int bi = 0;
    for (int s = 0; s < 16; ++s) {
      const float v = red_v[tid][s];
      if (v > best) { best = v; bi = red_i[tid][s]; }
    }
    idx_out[rowbase + tid] = bi;
  }
}

// ---------------------------------------------------------------------------
// Pre-convert a codebook to bf16, padding row length D up to Dp with zeros.
// ---------------------------------------------------------------------------
__global__ __launch_bounds__(256)
void cvt_bf16_kernel(const float* __restrict__ src, int D, int Dp, int total,
                     unsigned short* __restrict__ dst) {
  const int e = blockIdx.x * 256 + threadIdx.x;   // over K*Dp
  if (e >= total) return;
  const int row = e / Dp, k = e - row * Dp;
  dst[e] = f2bf((k < D) ? src[(long)row * D + k] : 0.0f);
}

// ---------------------------------------------------------------------------
// Small scalar kernels
// ---------------------------------------------------------------------------
__global__ __launch_bounds__(1024)
void weights_vq_kernel(const float* __restrict__ dw_w1, const float* __restrict__ pw_w1,
                       const float* __restrict__ dw_w2, const float* __restrict__ pw_w2,
                       const float* __restrict__ cb_dw, const float* __restrict__ cb_pw,
                       const float* __restrict__ cb_dw2, const float* __restrict__ cb_pw2,
                       float* __restrict__ qw1, float* __restrict__ qpw1,
                       float* __restrict__ qw2, float* __restrict__ qpw2) {
  const int t = threadIdx.x;
  if (t < 32) {
    float w[9];
    for (int i = 0; i < 9; ++i) w[i] = dw_w1[t * 9 + i];
    float best = 3.4e38f; int bk = 0;
    for (int k = 0; k < 256; ++k) {
      float d = 0.f;
      for (int i = 0; i < 9; ++i) { float e = w[i] - cb_dw[k * 9 + i]; d += e * e; }
      if (d < best) { best = d; bk = k; }
    }
    for (int i = 0; i < 9; ++i) qw1[t * 9 + i] = cb_dw[bk * 9 + i];
  } else if (t < 64) {
    const int c = t - 32;
    float w[9];
    for (int i = 0; i < 9; ++i) w[i] = dw_w2[c * 9 + i];
    float best = 3.4e38f; int bk = 0;
    for (int k = 0; k < 256; ++k) {
      float d = 0.f;
      for (int i = 0; i < 9; ++i) { float e = w[i] - cb_dw2[k * 9 + i]; d += e * e; }
      if (d < best) { best = d; bk = k; }
    }
    for (int i = 0; i < 9; ++i) qw2[c * 9 + i] = cb_dw2[bk * 9 + i];
  }
  {
    const float w = pw_w1[t];
    float best = 3.4e38f; int bk = 0;
    for (int k = 0; k < 256; ++k) { float e = w - cb_pw[k]; e *= e; if (e < best) { best = e; bk = k; } }
    qpw1[t] = cb_pw[bk];
  }
  {
    const float w = pw_w2[t];
    float best = 3.4e38f; int bk = 0;
    for (int k = 0; k < 256; ++k) { float e = w - cb_pw2[k]; e *= e; if (e < best) { best = e; bk = k; } }
    qpw2[t] = cb_pw2[bk];
  }
}

__global__ __launch_bounds__(256)
void norm_kernel(const float* __restrict__ cb, int D, float* __restrict__ out) {
  __shared__ float red[256];
  const int j = blockIdx.x;
  float s = 0.f;
  for (int d = threadIdx.x; d < D; d += 256) { const float v = cb[(long)j * D + d]; s += v * v; }
  red[threadIdx.x] = s;
  __syncthreads();
  for (int w = 128; w > 0; w >>= 1) {
    if (threadIdx.x < w) red[threadIdx.x] += red[threadIdx.x + w];
    __syncthreads();
  }
  if (threadIdx.x == 0) out[j] = red[0];
}

// h2 = lrelu(dwconv(cb_fdw[idx1] as 48x48, qw1))  -> [512][46*46]
__global__ __launch_bounds__(256)
void conv1_kernel(const float* __restrict__ cb_fdw, const int* __restrict__ idx1,
                  const float* __restrict__ qw1, float* __restrict__ h2) {
  const int n = blockIdx.x;                // b*32+c
  const int c = n & 31;
  const float* in = cb_fdw + (long)idx1[n] * 2304;
  const float* f  = qw1 + c * 9;
  for (int p = threadIdx.x; p < 46 * 46; p += 256) {
    const int y = p / 46, x = p - y * 46;
    const float* s = in + y * 48 + x;
    float v = s[0]*f[0]  + s[1]*f[1]  + s[2]*f[2]
            + s[48]*f[3] + s[49]*f[4] + s[50]*f[5]
            + s[96]*f[6] + s[97]*f[7] + s[98]*f[8];
    h2[(long)n * 2116 + p] = (v >= 0.f) ? v : LEAK * v;
  }
}

// factorized VQ3: row (b,g,c) = qpw1[g*32+c] * cb_fpw[idx2[b*32+c]]
// d(j) = s^2*||v_i||^2 - 2 s G[i][j] + ||u_j||^2
__global__ __launch_bounds__(256)
void idx3_kernel(const int* __restrict__ idx2, const float* __restrict__ qpw1,
                 const float* __restrict__ G, const float* __restrict__ pnorm,
                 const float* __restrict__ qnorm, int* __restrict__ idx3) {
  const int n = blockIdx.x * 256 + threadIdx.x;   // 16384
  const int b = n >> 10, g = (n >> 5) & 31, c = n & 31;
  const int i = idx2[b * 32 + c];
  const float s = qpw1[g * 32 + c];
  const float* Gi = G + (long)i * KCB;
  const float base = s * s * pnorm[i];
  float best = 3.4e38f; int bj = 0;
  for (int j = 0; j < KCB; ++j) {
    const float d = base - 2.f * s * Gi[j] + qnorm[j];
    if (d < best) { best = d; bj = j; }
  }
  idx3[n] = bj;
}

// Ssum[b][g][p] = sum_c cb_fpw2[idx4[b,g,c]][p]
__global__ __launch_bounds__(256)
void ssum_kernel(const float* __restrict__ cb_fpw2, const int* __restrict__ idx4,
                 float* __restrict__ Ssum) {
  const int bg = blockIdx.x;               // b*32+g
  const int* id = idx4 + bg * 32;
  for (int p = threadIdx.x; p < 1936; p += 256) {
    float s = 0.f;
    for (int c = 0; c < 32; ++c) s += cb_fpw2[(long)id[c] * 1936 + p];
    Ssum[(long)bg * 1936 + p] = s;
  }
}

// out[b][o][p] = lrelu(BN(sum_g qpw2[o*32+g] * Ssum[b][g][p]))
__global__ __launch_bounds__(256)
void final_kernel(const float* __restrict__ Ssum, const float* __restrict__ qpw2,
                  const float* __restrict__ gamma, const float* __restrict__ beta,
                  const float* __restrict__ mean, const float* __restrict__ var,
                  float* __restrict__ out) {
  const int e = blockIdx.x * 256 + threadIdx.x;
  if (e >= 16 * 32 * 1936) return;
  const int p = e % 1936;
  const int o = (e / 1936) & 31;
  const int b = e / (1936 * 32);
  const float* S = Ssum + ((long)b * 32) * 1936 + p;
  const float* w = qpw2 + o * 32;
  float acc = 0.f;
  for (int g = 0; g < 32; ++g) acc += w[g] * S[(long)g * 1936];
  float h = (acc - mean[o]) * rsqrtf(var[o] + BN_EPS) * gamma[o] + beta[o];
  out[e] = (h >= 0.f) ? h : LEAK * h;
}

// ---------------------------------------------------------------------------
extern "C" void kernel_launch(void* const* d_in, const int* in_sizes, int n_in,
                              void* d_out, int out_size, void* d_ws, size_t ws_size,
                              hipStream_t stream) {
  (void)in_sizes; (void)n_in; (void)out_size; (void)ws_size;
  const float* x       = (const float*)d_in[0];
  const float* dw_w1   = (const float*)d_in[1];
  const float* pw_w1   = (const float*)d_in[2];
  const float* dw_w2   = (const float*)d_in[3];
  const float* pw_w2   = (const float*)d_in[4];
  const float* cb_dw   = (const float*)d_in[5];
  const float* cb_pw   = (const float*)d_in[6];
  const float* cb_dw2  = (const float*)d_in[7];
  const float* cb_pw2  = (const float*)d_in[8];
  const float* cb_fdw  = (const float*)d_in[9];
  const float* cb_fpw  = (const float*)d_in[10];
  const float* cb_fdw2 = (const float*)d_in[11];
  const float* cb_fpw2 = (const float*)d_in[12];
  const float* gamma   = (const float*)d_in[13];
  const float* beta    = (const float*)d_in[14];
  const float* rmean   = (const float*)d_in[15];
  const float* rvar    = (const float*)d_in[16];
  float* out = (float*)d_out;

  float* w = (float*)d_ws;
  float* qw1    = w; w += 288;
  float* qw2    = w; w += 288;
  float* qpw1   = w; w += 1024;
  float* qpw2   = w; w += 1024;
  float* n_fdw  = w; w += 512;
  float* n_fpw  = w; w += 512;   // pnorm
  float* n_fdw2 = w; w += 512;   // qnorm
  float* n_fpw2 = w; w += 512;
  float* G      = w; w += 512 * 512;
  float* h2     = w; w += 512 * 2116;
  float* Ssum   = w; w += 512 * 1936;
  int* idx1 = (int*)w; w += 512;
  int* idx2 = (int*)w; w += 512;
  int* idx3 = (int*)w; w += 16384;
  int* idx4 = (int*)w; w += 16384;
  // padded bf16 codebooks (16B-aligned: all prior counts are multiples of 4)
  unsigned short* bf_fdw  = (unsigned short*)w;      // 512 x 2304
  unsigned short* bf_fpw  = bf_fdw  + 512 * 2304;    // 512 x 2144 (pad of 2116)
  unsigned short* bf_fdw2 = bf_fpw  + 512 * 2144;    // 512 x 2144
  unsigned short* bf_fpw2 = bf_fdw2 + 512 * 2144;    // 512 x 1952 (pad of 1936)

  weights_vq_kernel<<<1, 1024, 0, stream>>>(dw_w1, pw_w1, dw_w2, pw_w2,
                                            cb_dw, cb_pw, cb_dw2, cb_pw2,
                                            qw1, qpw1, qw2, qpw2);
  norm_kernel<<<512, 256, 0, stream>>>(cb_fdw,  2304, n_fdw);
  norm_kernel<<<512, 256, 0, stream>>>(cb_fpw,  2116, n_fpw);
  norm_kernel<<<512, 256, 0, stream>>>(cb_fdw2, 2116, n_fdw2);
  norm_kernel<<<512, 256, 0, stream>>>(cb_fpw2, 1936, n_fpw2);
  cvt_bf16_kernel<<<(512*2304 + 255)/256, 256, 0, stream>>>(cb_fdw,  2304, 2304, 512*2304, bf_fdw);
  cvt_bf16_kernel<<<(512*2144 + 255)/256, 256, 0, stream>>>(cb_fpw,  2116, 2144, 512*2144, bf_fpw);
  cvt_bf16_kernel<<<(512*2144 + 255)/256, 256, 0, stream>>>(cb_fdw2, 2116, 2144, 512*2144, bf_fdw2);
  cvt_bf16_kernel<<<(512*1952 + 255)/256, 256, 0, stream>>>(cb_fpw2, 1936, 1952, 512*1952, bf_fpw2);

  // VQ1: x rows vs cb_fdw
  vq_gemm_kernel<<<32, 256, 0, stream>>>(x, 2304, bf_fdw, 2304, n_fdw,
                                         512, 2304, idx1, nullptr);
  // conv1 + lrelu on gathered codewords
  conv1_kernel<<<512, 256, 0, stream>>>(cb_fdw, idx1, qw1, h2);
  // VQ2: h2 rows vs cb_fpw
  vq_gemm_kernel<<<32, 256, 0, stream>>>(h2, 2116, bf_fpw, 2144, n_fpw,
                                         512, 2116, idx2, nullptr);
  // Gram matrix G = cb_fpw . cb_fdw2^T  (enables factorized VQ3)
  vq_gemm_kernel<<<32, 256, 0, stream>>>(cb_fpw, 2116, bf_fdw2, 2144, nullptr,
                                         512, 2116, nullptr, G);
  // factorized VQ3 argmin
  idx3_kernel<<<64, 256, 0, stream>>>(idx2, qpw1, G, n_fpw, n_fdw2, idx3);
  // fused conv2 + VQ4
  conv_vq_kernel<<<1024, 256, 0, stream>>>(cb_fdw2, idx3, qw2, bf_fpw2, n_fpw2, idx4);
  // channel sum then mix + BN + lrelu
  ssum_kernel<<<512, 256, 0, stream>>>(cb_fpw2, idx4, Ssum);
  final_kernel<<<(16 * 32 * 1936 + 255) / 256, 256, 0, stream>>>(
      Ssum, qpw2, gamma, beta, rmean, rvar, out);
}
